// GNN4Contrastive_35261681500246
// MI455X (gfx1250) — matrix operand order repaired and checked by reading, not compiled, not run on hardware
//
#include <hip/hip_runtime.h>
#include <stdint.h>

typedef __attribute__((ext_vector_type(16))) __bf16 v16bf;
typedef __attribute__((ext_vector_type(8)))  float  v8f;

#define FT 128
#define LEAK 0.2f

union FragAB { uint4 u[2]; v16bf v; };

// ---------- order-preserving float <-> uint encoding for atomic max ----------
__device__ __forceinline__ unsigned enc_f(float f) {
  unsigned b = __float_as_uint(f);
  return (b & 0x80000000u) ? ~b : (b | 0x80000000u);
}
__device__ __forceinline__ float dec_f(unsigned u) {
  unsigned b = (u & 0x80000000u) ? (u ^ 0x80000000u) : ~u;
  return __uint_as_float(b);
}

// ---------- conversions ----------
__global__ void k_convert_w(const float* __restrict__ W, __bf16* __restrict__ Wt) {
  int idx = blockIdx.x * 256 + threadIdx.x;             // FT*FT = 16384
  if (idx < FT * FT) {
    int k = idx / FT, c = idx % FT;
    Wt[c * FT + k] = (__bf16)W[idx];                    // Wt[c][k] = W[k][c]
  }
}

__global__ void k_convert_x(const float* __restrict__ x, __bf16* __restrict__ xb, int total) {
  int idx = blockIdx.x * 256 + threadIdx.x;
  if (idx < total) xb[idx] = (__bf16)x[idx];
}

// ---------- WMMA GEMM: h[n x 128] = xb[n x 128] @ W (Wt is K-contiguous per column) ----------
__global__ void __launch_bounds__(256)
k_gemm_bf16(const __bf16* __restrict__ xb, const __bf16* __restrict__ Wt,
            float* __restrict__ h, int nrows) {
  __shared__ __align__(16) __bf16 ldsW[FT * FT];        // 32 KB
  {
    const uint4* s = (const uint4*)Wt;
    uint4* d = (uint4*)ldsW;
    for (int i = threadIdx.x; i < FT * FT / 8; i += 256) d[i] = s[i];
  }
  __syncthreads();

  const int wave = threadIdx.x >> 5;
  const int lane = threadIdx.x & 31;
  const int half = lane & 15;
  const int hi   = lane >> 4;
  const int rowBase = (blockIdx.x * 8 + wave) * 16;

  int arow = rowBase + half;
  if (arow >= nrows) arow = nrows - 1;                  // clamp (stores guarded)
  const __bf16* arowp = xb + (long)arow * FT;

  v8f acc[8];
#pragma unroll
  for (int ct = 0; ct < 8; ++ct)
#pragma unroll
    for (int j = 0; j < 8; ++j) acc[ct][j] = 0.0f;

#pragma unroll
  for (int kb = 0; kb < FT; kb += 32) {
    // A 16x32 bf16 frag (ISA 7.12.2): lanes<16 -> K {kb..kb+7, kb+16..kb+23},
    // lanes>=16 -> K {kb+8..kb+15, kb+24..kb+31}; row M = lane&15.
    FragAB a;
    a.u[0] = *(const uint4*)(arowp + kb + hi * 8);
    a.u[1] = *(const uint4*)(arowp + kb + 16 + hi * 8);
#pragma unroll
    for (int ct = 0; ct < 8; ++ct) {
      // B 32x16 bf16 frag: col N = lane&15; lanes<16 -> K {kb..kb+7, kb+16..kb+23}, lanes>=16 -> K {kb+8..kb+15, kb+24..kb+31}.
      FragAB b;
      const __bf16* bp = &ldsW[(ct * 16 + half) * FT + kb + hi * 8];
      b.u[0] = *(const uint4*)(bp);
      b.u[1] = *(const uint4*)(bp + 16);
      acc[ct] = __builtin_amdgcn_wmma_f32_16x16x32_bf16(
          false, a.v, false, b.v, (short)0, acc[ct], false, false);
    }
  }

  // C/D layout: VGPR j -> M = j + hi*8, col = lane&15 within the 16-wide tile.
#pragma unroll
  for (int ct = 0; ct < 8; ++ct) {
#pragma unroll
    for (int j = 0; j < 8; ++j) {
      int r = rowBase + j + hi * 8;
      if (r < nrows) h[(long)r * FT + ct * 16 + half] = acc[ct][j];
    }
  }
}

// ---------- per-node attention scores: ss = h . a_src, sd = h . a_dst ----------
__global__ void k_scores(const float* __restrict__ h, const float* __restrict__ a_src,
                         const float* __restrict__ a_dst, float* __restrict__ ss,
                         float* __restrict__ sd, int n) {
  int t = blockIdx.x * 256 + threadIdx.x;
  int node = t >> 2;
  int q = t & 3;
  if (node >= n) return;
  const float* hr = h + (long)node * FT + q * 32;
  float ps = 0.f, pd = 0.f;
#pragma unroll
  for (int i = 0; i < 32; ++i) {
    float v = hr[i];
    ps += v * a_src[q * 32 + i];
    pd += v * a_dst[q * 32 + i];
  }
  ps += __shfl_xor(ps, 1, 32); pd += __shfl_xor(pd, 1, 32);
  ps += __shfl_xor(ps, 2, 32); pd += __shfl_xor(pd, 2, 32);
  if (q == 0) { ss[node] = ps; sd[node] = pd; }
}

// ---------- zero the per-layer state ----------
__global__ void k_init(float* __restrict__ acc, unsigned* __restrict__ menc,
                       float* __restrict__ denom, int n, int total) {
  int idx = blockIdx.x * 256 + threadIdx.x;
  if (idx < total) acc[idx] = 0.0f;
  if (idx < n) { menc[idx] = 0u; denom[idx] = 0.0f; }
}

// ---------- segment max of leaky-relu edge scores ----------
__global__ void k_edge_max(const int* __restrict__ ei, const float* __restrict__ ss,
                           const float* __restrict__ sd, unsigned* __restrict__ menc,
                           int e, int n) {
  int i = blockIdx.x * 256 + threadIdx.x;
  if (i >= e + n) return;
  int s, d;
  if (i < e) { s = ei[i]; d = ei[e + i]; } else { s = d = i - e; }   // self loops
  float v = ss[s] + sd[d];
  v = v < 0.f ? v * LEAK : v;
  atomicMax(menc + d, enc_f(v));
}

// ---------- segment sum of exp(e - m) ----------
__global__ void k_edge_expsum(const int* __restrict__ ei, const float* __restrict__ ss,
                              const float* __restrict__ sd, const unsigned* __restrict__ menc,
                              float* __restrict__ denom, int e, int n) {
  int i = blockIdx.x * 256 + threadIdx.x;
  if (i >= e + n) return;
  int s, d;
  if (i < e) { s = ei[i]; d = ei[e + i]; } else { s = d = i - e; }
  float v = ss[s] + sd[d];
  v = v < 0.f ? v * LEAK : v;
  atomicAdd(denom + d, __expf(v - dec_f(menc[d])));
}

// ---------- aggregate: acc[dst] += alpha * h[src]   (one wave per edge) ----------
__global__ void __launch_bounds__(256)
k_aggregate(const int* __restrict__ ei, const float* __restrict__ ss,
            const float* __restrict__ sd, const unsigned* __restrict__ menc,
            const float* __restrict__ denom, const float* __restrict__ h,
            float* __restrict__ acc, int e, int n) {
  int wedge = blockIdx.x * 8 + (threadIdx.x >> 5);
  int lane  = threadIdx.x & 31;
  if (wedge >= e + n) return;
  int s, d;
  if (wedge < e) { s = ei[wedge]; d = ei[e + wedge]; } else { s = d = wedge - e; }
  float v = ss[s] + sd[d];
  v = v < 0.f ? v * LEAK : v;
  float alpha = __expf(v - dec_f(menc[d])) / denom[d];
  const float4 hv = *(const float4*)(h + (long)s * FT + lane * 4);
  float* out = acc + (long)d * FT + lane * 4;
  atomicAdd(out + 0, alpha * hv.x);
  atomicAdd(out + 1, alpha * hv.y);
  atomicAdd(out + 2, alpha * hv.z);
  atomicAdd(out + 3, alpha * hv.w);
}

// ---------- layer-1 epilogue: x1 = tanh(acc + b1); also recast to bf16 for layer 2 ----------
__global__ void k_epilogue1(const float* __restrict__ acc, const float* __restrict__ b,
                            float* __restrict__ x1, __bf16* __restrict__ xb, int total) {
  int idx = blockIdx.x * 256 + threadIdx.x;
  if (idx >= total) return;
  float v = tanhf(acc[idx] + b[idx & (FT - 1)]);
  x1[idx] = v;
  xb[idx] = (__bf16)v;
}

// ---------- final: out = max(x1, acc2 + b2)   (x1 lives in d_out) ----------
__global__ void k_final(const float* __restrict__ acc, const float* __restrict__ b,
                        float* __restrict__ out, int total) {
  int idx = blockIdx.x * 256 + threadIdx.x;
  if (idx >= total) return;
  float v = acc[idx] + b[idx & (FT - 1)];
  out[idx] = fmaxf(out[idx], v);
}

static inline size_t align256(size_t x) { return (x + 255) & ~(size_t)255; }

extern "C" void kernel_launch(void* const* d_in, const int* in_sizes, int n_in,
                              void* d_out, int out_size, void* d_ws, size_t ws_size,
                              hipStream_t stream) {
  const float* x      = (const float*)d_in[0];
  const int*   ei     = (const int*)d_in[1];   // [2, E]: row0 = src, row1 = dst
  const float* W1     = (const float*)d_in[3];
  const float* a_src1 = (const float*)d_in[4];
  const float* a_dst1 = (const float*)d_in[5];
  const float* b1     = (const float*)d_in[6];
  const float* W2     = (const float*)d_in[7];
  const float* a_src2 = (const float*)d_in[8];
  const float* a_dst2 = (const float*)d_in[9];
  const float* b2     = (const float*)d_in[10];

  const int n  = in_sizes[0] / FT;   // 100000
  const int e  = in_sizes[2];        // 1600000 (edge_weight length)
  const int nf = n * FT;             // 12.8M elements

  // workspace carve-out
  char* p = (char*)d_ws;
  float*    h     = (float*)p;                 p += align256((size_t)nf * 4);
  float*    acc   = (float*)p;                 p += align256((size_t)nf * 4);
  __bf16*   xb    = (__bf16*)p;                p += align256((size_t)nf * 2);
  __bf16*   w1t   = (__bf16*)p;                p += align256((size_t)FT * FT * 2);
  __bf16*   w2t   = (__bf16*)p;                p += align256((size_t)FT * FT * 2);
  float*    ss    = (float*)p;                 p += align256((size_t)n * 4);
  float*    sd    = (float*)p;                 p += align256((size_t)n * 4);
  unsigned* menc  = (unsigned*)p;              p += align256((size_t)n * 4);
  float*    denom = (float*)p;                 p += align256((size_t)n * 4);

  float* out = (float*)d_out;  // doubles as x1 storage between layers

  const int gW    = (FT * FT + 255) / 256;
  const int gNF   = (nf + 255) / 256;
  const int gRow  = (n + 127) / 128;           // 8 waves x 16 rows per block
  const int gSc   = (n * 4 + 255) / 256;
  const int gEdge = (e + n + 255) / 256;
  const int gAgg  = (e + n + 7) / 8;           // one wave per edge

  // ---- shared preprocessing ----
  k_convert_w<<<gW, 256, 0, stream>>>(W1, w1t);
  k_convert_w<<<gW, 256, 0, stream>>>(W2, w2t);
  k_convert_x<<<gNF, 256, 0, stream>>>(x, xb, nf);

  // ---- layer 1 ----
  k_gemm_bf16<<<gRow, 256, 0, stream>>>(xb, w1t, h, n);
  k_scores<<<gSc, 256, 0, stream>>>(h, a_src1, a_dst1, ss, sd, n);
  k_init<<<gNF, 256, 0, stream>>>(acc, menc, denom, n, nf);
  k_edge_max<<<gEdge, 256, 0, stream>>>(ei, ss, sd, menc, e, n);
  k_edge_expsum<<<gEdge, 256, 0, stream>>>(ei, ss, sd, menc, denom, e, n);
  k_aggregate<<<gAgg, 256, 0, stream>>>(ei, ss, sd, menc, denom, h, acc, e, n);
  k_epilogue1<<<gNF, 256, 0, stream>>>(acc, b1, out, xb, nf);   // x1 -> d_out (+bf16 copy)

  // ---- layer 2 ----
  k_gemm_bf16<<<gRow, 256, 0, stream>>>(xb, w2t, h, n);
  k_scores<<<gSc, 256, 0, stream>>>(h, a_src2, a_dst2, ss, sd, n);
  k_init<<<gNF, 256, 0, stream>>>(acc, menc, denom, n, nf);
  k_edge_max<<<gEdge, 256, 0, stream>>>(ei, ss, sd, menc, e, n);
  k_edge_expsum<<<gEdge, 256, 0, stream>>>(ei, ss, sd, menc, denom, e, n);
  k_aggregate<<<gAgg, 256, 0, stream>>>(ei, ss, sd, menc, denom, h, acc, e, n);
  k_final<<<gNF, 256, 0, stream>>>(acc, b2, out, nf);
}
